// GATv2block_29618094474250
// MI455X (gfx1250) — matrix operand order repaired and checked by reading, not compiled, not run on hardware
//
#include <hip/hip_runtime.h>

// ---------------------------------------------------------------------------
// CDNA5 (gfx1250) wave32 WMMA implementation of the GATv2 transformer block.
// Matrix ops use v_wmma_f32_16x16x32_bf16 with on-the-fly f32->bf16 operand
// conversion (memory-bound workload: keep weights single-pass from HBM).
// lin_W is staged to LDS with gfx1250 async global->LDS copies (ASYNCcnt).
// ---------------------------------------------------------------------------

typedef __attribute__((ext_vector_type(16))) __bf16 v16bf;
typedef __attribute__((ext_vector_type(8)))  float  v8f;

constexpr int B    = 8;
constexpr int G    = 3000;
constexpr int F    = 64;
constexpr int H    = 4;
constexpr int E    = 24000;
constexpr int DFF  = 100;
constexpr int NBLK = 2;     // NB
constexpr int NTRI = 2;     // NT
constexpr int ETOT = E + G; // edges + self loops
constexpr int NHF  = H * F; // 256
constexpr int CAT  = NTRI * F; // 128

#define NEG_SLOPE 0.2f
#define LN_EPS    1e-5f

// A-fragment K map (16-bit A 16x32): lanes 0-15 K=0..15, lanes 16-31 K=16..31
__device__ __forceinline__ int kmapA(int e, int half) {
  return (half << 4) + e;
}
// B-fragment K map (16-bit B 32x16): lanes 0-15 K=0..15, lanes 16-31 K=16..31
__device__ __forceinline__ int kmapB(int e, int half) {
  return (half << 4) + e;
}

__device__ __forceinline__ void atomicMaxF(float* addr, float val) {
  if (val >= 0.f) atomicMax((int*)addr, __float_as_int(val));
  else            atomicMin((unsigned int*)addr, __float_as_uint(val));
}

// ---------------------------------------------------------------------------
__global__ void k_fill(float* __restrict__ p, float v, int n) {
  int i = blockIdx.x * blockDim.x + threadIdx.x;
  if (i < n) p[i] = v;
}

// ---------------------------------------------------------------------------
// xl/xr transform: out[M,256] = A[M,64] * W[64,256] + bias[256]
// One wave per 16-row stripe x 4 N-tiles; A fragments loaded once, 8 WMMAs.
__global__ void k_xform(const float* __restrict__ A, const float* __restrict__ W,
                        const float* __restrict__ bias, float* __restrict__ out,
                        int Mtiles) {
  int lane = threadIdx.x & 31;
  int tile = blockIdx.x * (blockDim.x >> 5) + (threadIdx.x >> 5);
  if (tile >= Mtiles * 4) return;
  int ntg = (tile & 3) * 4;     // 4 N-tiles per wave
  int mt  = tile >> 2;
  int half = lane >> 4, l16 = lane & 15;
  const float* arow = A + (size_t)(mt * 16 + l16) * F;

  v16bf a0, a1;
#pragma unroll
  for (int e = 0; e < 16; ++e) {
    int ka = kmapA(e, half);
    a0[e] = (__bf16)arow[ka];
    a1[e] = (__bf16)arow[ka + 32];
  }

  for (int q = 0; q < 4; ++q) {
    int n = (ntg + q) * 16 + l16;
    v16bf b0, b1;
#pragma unroll
    for (int e = 0; e < 16; ++e) {
      int kb = kmapB(e, half);
      b0[e] = (__bf16)W[kb * NHF + n];
      b1[e] = (__bf16)W[(kb + 32) * NHF + n];
    }
    v8f c = {};
    c = __builtin_amdgcn_wmma_f32_16x16x32_bf16(false, a0, false, b0, (short)0, c, false, false);
    c = __builtin_amdgcn_wmma_f32_16x16x32_bf16(false, a1, false, b1, (short)0, c, false, false);
    float bv = bias[n];
#pragma unroll
    for (int r = 0; r < 8; ++r) {
      int m = mt * 16 + half * 8 + r;
      out[(size_t)m * NHF + n] = c[r] + bv;
    }
  }
}

// ---------------------------------------------------------------------------
// Per-edge logits + running max (scatter-softmax pass 1)
__global__ void k_logits(const float* __restrict__ xl, const float* __restrict__ xr,
                         const int* __restrict__ sidx, const int* __restrict__ didx,
                         const float* __restrict__ att, float* __restrict__ logits,
                         float* __restrict__ mbuf) {
  int t = blockIdx.x * blockDim.x + threadIdx.x;
  if (t >= B * ETOT * H) return;
  int h = t & (H - 1);
  int e = (t >> 2) % ETOT;
  int b = t / (ETOT * H);
  int s  = (e < E) ? sidx[e] : (e - E);
  int dd = (e < E) ? didx[e] : (e - E);
  const float4* pl = (const float4*)(xl + (((size_t)b * G + s) * H + h) * F);
  const float4* pr = (const float4*)(xr + (((size_t)b * G + dd) * H + h) * F);
  const float4* pa = (const float4*)(att + (size_t)h * F);
  float acc = 0.f;
#pragma unroll 4
  for (int k = 0; k < F / 4; ++k) {
    float4 l = pl[k], r = pr[k], w = pa[k];
    float v;
    v = l.x + r.x; acc += (v > 0.f ? v : NEG_SLOPE * v) * w.x;
    v = l.y + r.y; acc += (v > 0.f ? v : NEG_SLOPE * v) * w.y;
    v = l.z + r.z; acc += (v > 0.f ? v : NEG_SLOPE * v) * w.z;
    v = l.w + r.w; acc += (v > 0.f ? v : NEG_SLOPE * v) * w.w;
  }
  logits[t] = acc;
  atomicMaxF(&mbuf[((size_t)b * G + dd) * H + h], acc);
}

// exp + denominator (pass 2)
__global__ void k_expsum(const int* __restrict__ didx, float* __restrict__ logits,
                         const float* __restrict__ mbuf, float* __restrict__ den) {
  int t = blockIdx.x * blockDim.x + threadIdx.x;
  if (t >= B * ETOT * H) return;
  int h = t & (H - 1);
  int e = (t >> 2) % ETOT;
  int b = t / (ETOT * H);
  int dd = (e < E) ? didx[e] : (e - E);
  float ex = expf(logits[t] - mbuf[((size_t)b * G + dd) * H + h]);
  logits[t] = ex;
  atomicAdd(&den[((size_t)b * G + dd) * H + h], ex);
}

// weighted message scatter-add (pass 3)
__global__ void k_scatter(const float* __restrict__ xl, const int* __restrict__ sidx,
                          const int* __restrict__ didx, const float* __restrict__ ex,
                          const float* __restrict__ den, float* __restrict__ accm) {
  int t = blockIdx.x * blockDim.x + threadIdx.x;
  if (t >= B * ETOT * H) return;
  int h = t & (H - 1);
  int e = (t >> 2) % ETOT;
  int b = t / (ETOT * H);
  int s  = (e < E) ? sidx[e] : (e - E);
  int dd = (e < E) ? didx[e] : (e - E);
  float alpha = ex[t] / den[((size_t)b * G + dd) * H + h];
  const float* pl = xl + (((size_t)b * G + s) * H + h) * F;
  float* po = accm + (((size_t)b * G + dd) * H + h) * F;
#pragma unroll 8
  for (int k = 0; k < F; ++k) atomicAdd(&po[k], alpha * pl[k]);
}

// head mean + gat bias -> slot j of x_cat
__global__ void k_headmean(const float* __restrict__ accm, const float* __restrict__ bias,
                           float* __restrict__ xcat, int jslot) {
  int t = blockIdx.x * blockDim.x + threadIdx.x;
  if (t >= B * G * F) return;
  int f = t & (F - 1);
  int g = (t >> 6) % G;
  int b = t / (G * F);
  size_t base = ((size_t)b * G + g) * H * F;
  float v = 0.25f * (accm[base + f] + accm[base + F + f] +
                     accm[base + 2 * F + f] + accm[base + 3 * F + f]) + bias[f];
  xcat[((size_t)b * G + g) * CAT + jslot * F + f] = v;
}

// ---------------------------------------------------------------------------
// Per-gene linear: out[b,g,f] = sum_c xcat[b,g,c] * linW[g,c,f] + linb[g,f]
// One block = 2 genes. The 64KB weight slab for both genes is staged into LDS
// with gfx1250 async global->LDS copies, then 8 waves (gene x n-tile) consume
// it through DS reads into WMMA B-fragments. M=8 rows (rows 8..15 masked).
__global__ void __launch_bounds__(256) k_lin(const float* __restrict__ xcat,
                                             const float* __restrict__ Wall,
                                             const float* __restrict__ ball,
                                             float* __restrict__ out) {
  __shared__ __align__(16) float ldsW[2 * CAT * F];  // 64 KB
  int g0 = blockIdx.x * 2;
  const float* src = Wall + (size_t)g0 * CAT * F;

  // Cooperative async stage: 2*128*64 floats = 16 iters * 256 thr * 16B
#pragma unroll 4
  for (int it = 0; it < 16; ++it) {
    int idx = (it * 256 + threadIdx.x) * 4;
    unsigned lo = (unsigned)(size_t)&ldsW[idx];
    const float* gp = src + idx;
    asm volatile("global_load_async_to_lds_b128 %0, %1, off"
                 :: "v"(lo), "v"(gp) : "memory");
  }
  asm volatile("s_wait_asynccnt 0x0" ::: "memory");
  __syncthreads();

  int lane = threadIdx.x & 31;
  int w    = threadIdx.x >> 5;
  int lg   = w >> 2;          // which of the 2 genes
  int nt   = w & 3;           // n-tile
  int g    = g0 + lg;
  int half = lane >> 4, l16 = lane & 15;
  int n = nt * 16 + l16;
  const float* Wg = ldsW + lg * CAT * F;

  v8f c = {};
#pragma unroll
  for (int kc = 0; kc < 4; ++kc) {
    int kb = kc * 32;
    v16bf a, bfr;
#pragma unroll
    for (int e = 0; e < 16; ++e) {
      int ka = kb + kmapA(e, half);
      int kB = kb + kmapB(e, half);
      float av = (l16 < 8) ? xcat[((size_t)l16 * G + g) * CAT + ka] : 0.f;
      a[e]   = (__bf16)av;
      bfr[e] = (__bf16)Wg[kB * F + n];
    }
    c = __builtin_amdgcn_wmma_f32_16x16x32_bf16(false, a, false, bfr, (short)0, c, false, false);
  }
  if (half == 0) {
    float bv = ball[(size_t)g * F + n];
#pragma unroll
    for (int r = 0; r < 8; ++r)
      out[((size_t)r * G + g) * F + n] = c[r] + bv;
  }
}

// FF1: h[b,g,d] = relu(X[b,g,:]*W1[g,:,d] + b1[g,d]), D=100 (7 n-tiles, masked)
__global__ void k_ff1(const float* __restrict__ X, const float* __restrict__ Wall,
                      const float* __restrict__ ball, float* __restrict__ hout) {
  int lane = threadIdx.x & 31;
  int tile = blockIdx.x * (blockDim.x >> 5) + (threadIdx.x >> 5);
  if (tile >= G * 7) return;
  int nt = tile % 7, g = tile / 7;
  int half = lane >> 4, l16 = lane & 15;
  int n = nt * 16 + l16;
  bool nok = n < DFF;
  const float* Wg = Wall + (size_t)g * F * DFF;

  v8f c = {};
  for (int kc = 0; kc < 2; ++kc) {
    int kb = kc * 32;
    if (kc == 0) __builtin_prefetch(&Wg[32 * DFF], 0, 1); // global_prefetch
    v16bf a, bfr;
#pragma unroll
    for (int e = 0; e < 16; ++e) {
      int ka = kb + kmapA(e, half);
      int kB = kb + kmapB(e, half);
      float av = (l16 < 8) ? X[((size_t)l16 * G + g) * F + ka] : 0.f;
      float bv = nok ? Wg[kB * DFF + n] : 0.f;
      a[e]   = (__bf16)av;
      bfr[e] = (__bf16)bv;
    }
    c = __builtin_amdgcn_wmma_f32_16x16x32_bf16(false, a, false, bfr, (short)0, c, false, false);
  }
  if (half == 0 && nok) {
    float bv = ball[(size_t)g * DFF + n];
#pragma unroll
    for (int r = 0; r < 8; ++r)
      hout[((size_t)r * G + g) * DFF + n] = fmaxf(c[r] + bv, 0.f);
  }
}

// FF2: Xff[b,g,f] = h[b,g,:]*W2[g,:,f] + b2[g,f], K=100 (pad to 128, masked)
__global__ void k_ff2(const float* __restrict__ hin, const float* __restrict__ Wall,
                      const float* __restrict__ ball, float* __restrict__ out) {
  int lane = threadIdx.x & 31;
  int tile = blockIdx.x * (blockDim.x >> 5) + (threadIdx.x >> 5);
  if (tile >= G * 4) return;
  int nt = tile & 3, g = tile >> 2;
  int half = lane >> 4, l16 = lane & 15;
  int n = nt * 16 + l16;
  const float* Wg = Wall + (size_t)g * DFF * F;

  v8f c = {};
  for (int kc = 0; kc < 4; ++kc) {
    int kb = kc * 32;
    if (kc < 3) __builtin_prefetch(&Wg[(kb + 32) * F], 0, 1);
    v16bf a, bfr;
#pragma unroll
    for (int e = 0; e < 16; ++e) {
      int ka = kb + kmapA(e, half);
      int kB = kb + kmapB(e, half);
      float av = (l16 < 8 && ka < DFF) ? hin[((size_t)l16 * G + g) * DFF + ka] : 0.f;
      float bv = (kB < DFF) ? Wg[kB * F + n] : 0.f;
      a[e]   = (__bf16)av;
      bfr[e] = (__bf16)bv;
    }
    c = __builtin_amdgcn_wmma_f32_16x16x32_bf16(false, a, false, bfr, (short)0, c, false, false);
  }
  if (half == 0) {
    float bv = ball[(size_t)g * F + n];
#pragma unroll
    for (int r = 0; r < 8; ++r)
      out[((size_t)r * G + g) * F + n] = c[r] + bv;
  }
}

// LayerNorm(xres + delta) * gamma + beta -> out ; one wave per 64-elem row
__global__ void k_ln(const float* __restrict__ xres, const float* __restrict__ delta,
                     const float* __restrict__ gam, const float* __restrict__ bet,
                     float* __restrict__ out, int rows) {
  int lane = threadIdx.x & 31;
  int row = blockIdx.x * (blockDim.x >> 5) + (threadIdx.x >> 5);
  if (row >= rows) return;
  const float* a = xres + (size_t)row * F;
  const float* d = delta + (size_t)row * F;
  float x0 = a[lane] + d[lane];
  float x1 = a[lane + 32] + d[lane + 32];
  float s = x0 + x1, sq = x0 * x0 + x1 * x1;
#pragma unroll
  for (int off = 16; off; off >>= 1) {
    s  += __shfl_xor(s, off, 32);
    sq += __shfl_xor(sq, off, 32);
  }
  float mean = s * (1.f / F);
  float var  = sq * (1.f / F) - mean * mean;
  float inv  = rsqrtf(var + LN_EPS);
  float* o = out + (size_t)row * F;
  o[lane]      = (x0 - mean) * inv * gam[lane]      + bet[lane];
  o[lane + 32] = (x1 - mean) * inv * gam[lane + 32] + bet[lane + 32];
}

// ---------------------------------------------------------------------------
extern "C" void kernel_launch(void* const* d_in, const int* in_sizes, int n_in,
                              void* d_out, int out_size, void* d_ws, size_t ws_size,
                              hipStream_t stream) {
  (void)in_sizes; (void)n_in; (void)out_size; (void)ws_size;
  const float* X_input = (const float*)d_in[0];
  const int*   edges   = (const int*)  d_in[1];
  const float* Wl   = (const float*)d_in[2];
  const float* bl   = (const float*)d_in[3];
  const float* Wr   = (const float*)d_in[4];
  const float* br   = (const float*)d_in[5];
  const float* att  = (const float*)d_in[6];
  const float* gb   = (const float*)d_in[7];
  const float* linW = (const float*)d_in[8];
  const float* linb = (const float*)d_in[9];
  const float* f1W  = (const float*)d_in[10];
  const float* f1b  = (const float*)d_in[11];
  const float* f2W  = (const float*)d_in[12];
  const float* f2b  = (const float*)d_in[13];
  const float* ln1g = (const float*)d_in[14];
  const float* ln1b = (const float*)d_in[15];
  const float* ln2g = (const float*)d_in[16];
  const float* ln2b = (const float*)d_in[17];

  float* ws = (float*)d_ws;
  size_t off = 0;
  float* xl     = ws + off; off += (size_t)B * G * H * F;
  float* xr     = ws + off; off += (size_t)B * G * H * F;
  float* accm   = ws + off; off += (size_t)B * G * H * F;
  float* logits = ws + off; off += (size_t)B * ETOT * H;
  float* mbuf   = ws + off; off += (size_t)B * G * H;
  float* den    = ws + off; off += (size_t)B * G * H;
  float* xcat   = ws + off; off += (size_t)B * G * CAT;
  float* Xcur   = ws + off; off += (size_t)B * G * F;
  float* Xtmp   = ws + off; off += (size_t)B * G * F;
  float* hbuf   = ws + off; off += (size_t)B * G * DFF;

  hipMemcpyAsync(Xcur, X_input, sizeof(float) * B * G * F,
                 hipMemcpyDeviceToDevice, stream);

  const int TPB = 256;
  auto cdiv = [](long n, long d) { return (int)((n + d - 1) / d); };

  const int Mtiles   = (B * G) / 16;                 // 1500
  const int xformBlk = cdiv((long)Mtiles * 4, 8);    // 4 N-tiles per wave
  const int edgeBlk  = cdiv((long)B * ETOT * H, TPB);

  for (int i = 0; i < NBLK; ++i) {
    for (int j = 0; j < NTRI; ++j) {
      const int   ij   = i * NTRI + j;
      const int*  sidx = edges + (size_t)(j * 2 + 0) * E;
      const int*  didx = edges + (size_t)(j * 2 + 1) * E;

      k_fill<<<cdiv(B * G * H, TPB), TPB, 0, stream>>>(mbuf, -__builtin_huge_valf(), B * G * H);
      k_fill<<<cdiv(B * G * H, TPB), TPB, 0, stream>>>(den, 0.f, B * G * H);
      k_fill<<<cdiv((long)B * G * H * F, TPB), TPB, 0, stream>>>(accm, 0.f, B * G * H * F);

      k_xform<<<xformBlk, TPB, 0, stream>>>(Xcur, Wl + (size_t)ij * F * NHF,
                                            bl + (size_t)ij * NHF, xl, Mtiles);
      k_xform<<<xformBlk, TPB, 0, stream>>>(Xcur, Wr + (size_t)ij * F * NHF,
                                            br + (size_t)ij * NHF, xr, Mtiles);

      k_logits <<<edgeBlk, TPB, 0, stream>>>(xl, xr, sidx, didx,
                                             att + (size_t)ij * H * F, logits, mbuf);
      k_expsum <<<edgeBlk, TPB, 0, stream>>>(didx, logits, mbuf, den);
      k_scatter<<<edgeBlk, TPB, 0, stream>>>(xl, sidx, didx, logits, den, accm);
      k_headmean<<<cdiv((long)B * G * F, TPB), TPB, 0, stream>>>(accm, gb + (size_t)ij * F,
                                                                 xcat, j);
    }
    // per-gene linear (async LDS-staged weights) + LN1
    k_lin<<<G / 2, TPB, 0, stream>>>(xcat, linW, linb, Xtmp);
    k_ln <<<cdiv((long)B * G, 8), TPB, 0, stream>>>(Xcur, Xtmp, ln1g + (size_t)i * F,
                                                    ln1b + (size_t)i * F, Xcur, B * G);
    // FFN + LN2
    k_ff1<<<cdiv((long)G * 7, 8), TPB, 0, stream>>>(Xcur, f1W, f1b, hbuf);
    k_ff2<<<cdiv((long)G * 4, 8), TPB, 0, stream>>>(hbuf, f2W, f2b, Xtmp);
    k_ln <<<cdiv((long)B * G, 8), TPB, 0, stream>>>(Xcur, Xtmp, ln2g + (size_t)i * F,
                                                    ln2b + (size_t)i * F, Xcur, B * G);
  }

  hipMemcpyAsync(d_out, Xcur, sizeof(float) * B * G * F,
                 hipMemcpyDeviceToDevice, stream);
}
